// DotProductAttention_6674379178249
// MI455X (gfx1250) — matrix operand order repaired and checked by reading, not compiled, not run on hardware
//
#include <hip/hip_runtime.h>

#define B_  16
#define L_  2048
#define S_  2048
#define DK_ 128
#define DV_ 128
#define MT_ 32   // query rows per workgroup (two 16-row WMMA sub-tiles)

typedef __attribute__((ext_vector_type(16))) _Float16 v16h;
typedef __attribute__((ext_vector_type(8)))  float    v8f;
typedef __attribute__((ext_vector_type(4)))  float    v4f;

// K-offset of element pair (2v, 2v+1) in a 16-bit A fragment (16x32), per
// CDNA5 ISA 7.12.2: half 0 covers K {0-7,16-23}, half 1 {8-15,24-31}.
__device__ __forceinline__ int fragA_k(int v, int hh) {
  return ((v < 4) ? (2 * v) : (8 + 2 * v)) + 8 * hh;
}

__global__ __launch_bounds__(256) void attn_fused_kernel(
    const float* __restrict__ Q, const float* __restrict__ Km,
    const float* __restrict__ V, float* __restrict__ Out,
    float* __restrict__ Attn) {
  extern __shared__ float sP[];  // [MT_][S_] score/prob tile, 256 KB
  const int b    = blockIdx.y;
  const int row0 = blockIdx.x * MT_;
  const int tid  = threadIdx.x;
  const int wave = tid >> 5;
  const int lane = tid & 31;
  const int m    = lane & 15;   // A-row / B-col / D-col index for this lane
  const int hh   = lane >> 4;   // half-wave select
  const float scale = 0.08838834764831845f;  // 1/sqrt(128)

  // ---------- Phase 1: scores = scale * Q_tile @ K^T  -> LDS ----------
  // A fragments: two 16-row sub-tiles, DK=128 as 4 K-chunks of 32 each.
  v16h a[2][4];
#pragma unroll
  for (int ms = 0; ms < 2; ++ms) {
    const float* qrow = Q + ((size_t)b * L_ + row0 + ms * 16 + m) * DK_;
#pragma unroll
    for (int kc = 0; kc < 4; ++kc) {
#pragma unroll
      for (int v = 0; v < 8; ++v) {
        const int k = kc * 32 + fragA_k(v, hh);
        const float2 f = *(const float2*)(qrow + k);
        a[ms][kc][2 * v]     = (_Float16)f.x;
        a[ms][kc][2 * v + 1] = (_Float16)f.y;
      }
    }
  }

  for (int nt = wave; nt < S_ / 16; nt += 8) {
    const int col0 = nt * 16;
    // B fragment: score-tile column n (= m) is K-matrix row col0+n.
    const float* krow = Km + ((size_t)b * S_ + col0 + m) * DK_;
    v8f c0 = {}, c1 = {};
#pragma unroll
    for (int kc = 0; kc < 4; ++kc) {
      v16h bf;
      const float* p = krow + kc * 32 + 8 * hh;  // table K order: half 0 {0-7,16-23}, half 1 {8-15,24-31}
#pragma unroll
      for (int e = 0; e < 16; ++e) bf[e] = (_Float16)p[(e < 8) ? e : e + 8];
      // One B fragment feeds both M sub-tiles.
      c0 = __builtin_amdgcn_wmma_f32_16x16x32_f16(false, a[0][kc], false, bf,
                                                  (short)0, c0, false, false);
      c1 = __builtin_amdgcn_wmma_f32_16x16x32_f16(false, a[1][kc], false, bf,
                                                  (short)0, c1, false, false);
    }
#pragma unroll
    for (int r = 0; r < 8; ++r) {
      const int mr = hh * 8 + r;                  // D layout: m = 8*half + vgpr
      sP[(mr)      * S_ + col0 + m] = c0[r] * scale;
      sP[(mr + 16) * S_ + col0 + m] = c1[r] * scale;
    }
  }
  __syncthreads();

  // ---------- Phase 2: row-wise softmax in LDS (float4), stream Attn ----------
#pragma unroll
  for (int rr = 0; rr < 4; ++rr) {
    const int row = wave * 4 + rr;
    v4f* prow4 = (v4f*)(sP + row * S_);
    float mx = -3.402823466e+38f;
    for (int j = lane; j < S_ / 4; j += 32) {
      const v4f x = prow4[j];
      mx = fmaxf(mx, fmaxf(fmaxf(x[0], x[1]), fmaxf(x[2], x[3])));
    }
#pragma unroll
    for (int off = 16; off > 0; off >>= 1)
      mx = fmaxf(mx, __shfl_xor(mx, off, 32));
    float sum = 0.f;
    for (int j = lane; j < S_ / 4; j += 32) {
      const v4f x = prow4[j];
      v4f e;
      e[0] = __expf(x[0] - mx);
      e[1] = __expf(x[1] - mx);
      e[2] = __expf(x[2] - mx);
      e[3] = __expf(x[3] - mx);
      prow4[j] = e;
      sum += (e[0] + e[1]) + (e[2] + e[3]);
    }
#pragma unroll
    for (int off = 16; off > 0; off >>= 1)
      sum += __shfl_xor(sum, off, 32);
    const float inv = 1.0f / sum;
    v4f* arow4 = (v4f*)(Attn + ((size_t)b * L_ + row0 + row) * S_);
    for (int j = lane; j < S_ / 4; j += 32) {
      const v4f p = prow4[j] * inv;
      prow4[j] = p;                              // keep P resident for phase 3
      // attn is write-once and bigger than L2 (268MB > 192MB): stream it (NT)
      // so K/V stay L2-resident across the 64 workgroups per batch.
      __builtin_nontemporal_store(p, arow4 + j); // global_store_b128, 512B/wave
    }
  }
  __syncthreads();

  // ---------- Phase 3: out_tile = P @ V ----------
  const int n0 = wave * 16;                                  // 8 waves x 16 = DV
  const float* vcol = V + (size_t)b * S_ * DV_ + n0 + m;     // this lane's column
  v8f acc0 = {}, acc1 = {};
  for (int kt = 0; kt < S_ / 32; ++kt) {
    const int k0 = kt * 32;
    // B fragment from V (shared by both M sub-tiles).
    v16h bf;
#pragma unroll
    for (int e = 0; e < 16; ++e) {
      const int k = k0 + 8 * hh + ((e < 8) ? e : e + 8);
      bf[e] = (_Float16)vcol[(size_t)k * DV_];
    }
    // A fragments from LDS P for both sub-tiles.
#pragma unroll
    for (int ms = 0; ms < 2; ++ms) {
      v16h af;
#pragma unroll
      for (int v = 0; v < 8; ++v) {
        const int k = k0 + fragA_k(v, hh);
        const float2 f = *(const float2*)(sP + (ms * 16 + m) * S_ + k);
        af[2 * v]     = (_Float16)f.x;
        af[2 * v + 1] = (_Float16)f.y;
      }
      if (ms == 0)
        acc0 = __builtin_amdgcn_wmma_f32_16x16x32_f16(false, af, false, bf,
                                                      (short)0, acc0, false, false);
      else
        acc1 = __builtin_amdgcn_wmma_f32_16x16x32_f16(false, af, false, bf,
                                                      (short)0, acc1, false, false);
    }
  }
  float* orow = Out + ((size_t)b * L_ + row0) * DV_;
#pragma unroll
  for (int r = 0; r < 8; ++r) {
    const int mr = hh * 8 + r;
    __builtin_nontemporal_store(acc0[r], orow + (size_t)(mr)      * DV_ + n0 + m);
    __builtin_nontemporal_store(acc1[r], orow + (size_t)(mr + 16) * DV_ + n0 + m);
  }
}

extern "C" void kernel_launch(void* const* d_in, const int* in_sizes, int n_in,
                              void* d_out, int out_size, void* d_ws, size_t ws_size,
                              hipStream_t stream) {
  const float* Q = (const float*)d_in[0];
  const float* K = (const float*)d_in[1];
  const float* V = (const float*)d_in[2];
  float* Out  = (float*)d_out;                          // [B, L, DV]
  float* Attn = (float*)d_out + (size_t)B_ * L_ * DV_;  // [B, L, S]
  dim3 grid(L_ / MT_, B_);
  dim3 block(256);
  const size_t shmem = (size_t)MT_ * S_ * sizeof(float);  // 256 KB LDS
  hipLaunchKernelGGL(attn_fused_kernel, grid, block, shmem, stream,
                     Q, K, V, Out, Attn);
}
